// GNNEncoder_54614804136342
// MI455X (gfx1250) — matrix operand order repaired and checked by reading, not compiled, not run on hardware
//
#include <hip/hip_runtime.h>

// ---------------------------------------------------------------------------
// GCN encoder for MI455X (gfx1250, wave32).
// - GEMMs: v_wmma_f32_16x16x32_bf16, A-fragments from global (b128 loads),
//   B-fragments (W^T) from LDS, staged via global_load_async_to_lds_b128
//   (ASYNCcnt path, no VGPR round-trip), double-buffered ct-loop.
// - Aggregation: wave-per-edge, lane-per-channel coalesced global_atomic_add_f32
//   (node features are L2-resident: 25.6MB/layer vs 192MB L2).
// ---------------------------------------------------------------------------

typedef __attribute__((ext_vector_type(16))) __bf16 v16bf;
typedef __attribute__((ext_vector_type(8)))  __bf16 v8bf;
typedef __attribute__((ext_vector_type(8)))  float  v8f;

#define D_IN  256
#define D_HID 128
#define D_OUT 64
#define NUM_GRAPHS 16
#define BN_EPS 1e-5f

// ----------------------------- helpers -------------------------------------

__global__ __launch_bounds__(256) void k_to_bf16(const float* __restrict__ x,
                                                 __bf16* __restrict__ y,
                                                 long long total) {
  long long i = (long long)blockIdx.x * blockDim.x + threadIdx.x;
  if (i < total) y[i] = (__bf16)x[i];
}

// W [K, M] f32 row-major  ->  WT [M, K] bf16 row-major
__global__ __launch_bounds__(256) void k_transpose_bf16(const float* __restrict__ W,
                                                        __bf16* __restrict__ WT,
                                                        int K, int M) {
  int i = blockIdx.x * blockDim.x + threadIdx.x;
  if (i >= K * M) return;
  int k = i / M, m = i % M;
  WT[(long long)m * K + k] = (__bf16)W[i];
}

__global__ __launch_bounds__(256) void k_degree(const long long* __restrict__ dst,
                                                float* __restrict__ deg, int E) {
  int i = blockIdx.x * blockDim.x + threadIdx.x;
  if (i < E) atomicAdd(&deg[(int)dst[i]], 1.0f);
}

__global__ __launch_bounds__(256) void k_dinv(float* __restrict__ deg, int N) {
  int i = blockIdx.x * blockDim.x + threadIdx.x;
  if (i < N) deg[i] = rsqrtf(deg[i] + 1.0f);  // +1 self-loop, deg >= 1
}

// ----------------------------- WMMA GEMM -----------------------------------
// C[N, M] = A[N, K](bf16) @ W[K, M], with WT = W^T (bf16) staged in LDS via
// the CDNA5 async copy path (global_load_async_to_lds_b128, ASYNCcnt).
// Block: 256 threads = 8 waves; wave w computes rows [rt*16, rt*16+16) x all M.

template <int K, int M>
__global__ __launch_bounds__(256) void k_gemm_wmma(const __bf16* __restrict__ A,
                                                   const __bf16* __restrict__ WT,
                                                   float* __restrict__ C,
                                                   int n, int nRowTiles) {
  extern __shared__ char smem_raw[];
  __bf16* wt_lds = (__bf16*)smem_raw;                 // [M][K]

  const int tid = threadIdx.x;
  constexpr int TOT8 = (M * K) / 8;                   // 16B chunks
  for (int i = tid; i < TOT8; i += 256) {
    // LDS byte offset = low 32 bits of the flat shared-memory pointer
    unsigned lds_off = (unsigned)(size_t)(wt_lds + i * 8);
    const __bf16* gsrc = WT + i * 8;
    asm volatile("global_load_async_to_lds_b128 %0, %1, off"
                 :: "v"(lds_off), "v"(gsrc) : "memory");
  }
  asm volatile("s_wait_asynccnt 0x0" ::: "memory");   // my chunks landed
  __syncthreads();                                    // everyone's chunks landed

  const int wave = tid >> 5;
  const int lane = tid & 31;
  const int rt = blockIdx.x * 8 + wave;
  if (rt >= nRowTiles) return;

  const int row0   = rt * 16;
  const int laneLo = lane & 15;
  const int hiHalf = lane >> 4;                       // 0: lanes 0-15, 1: 16-31
  int aRow = row0 + laneLo; if (aRow >= n) aRow = n - 1;
  const __bf16* Abase = A + (long long)aRow * K + hiHalf * 16;

  constexpr int MT = M / 16;
  v8f acc[MT];
#pragma unroll
  for (int ct = 0; ct < MT; ++ct) acc[ct] = (v8f){0, 0, 0, 0, 0, 0, 0, 0};

#pragma unroll
  for (int k0 = 0; k0 < K; k0 += 32) {
    // A fragment (contiguous k, same order as B): lanes 0-15 hold K k0..k0+15,
    // lanes 16-31 hold K k0+16..+31.
    v8bf alo = *(const v8bf*)(Abase + k0);
    v8bf ahi = *(const v8bf*)(Abase + k0 + 8);
    v16bf afrag = __builtin_shufflevector(alo, ahi, 0, 1, 2, 3, 4, 5, 6, 7,
                                          8, 9, 10, 11, 12, 13, 14, 15);
    __builtin_prefetch(Abase + k0 + 64, 0, 1);        // global_prefetch_b8
    const int kb = k0 + hiHalf * 16;                  // B: upper half-wave K+16

    // Software-pipelined B fragments: issue ct+1 loads before ct's WMMA so
    // >=2 ds_load_b128 stay in flight and waits can be partial.
    const __bf16* bp0 = wt_lds + (long long)laneLo * K + kb;
    v8bf blo = *(const v8bf*)bp0;
    v8bf bhi = *(const v8bf*)(bp0 + 8);
#pragma unroll
    for (int ct = 0; ct < MT; ++ct) {
      v16bf bfrag = __builtin_shufflevector(blo, bhi, 0, 1, 2, 3, 4, 5, 6, 7,
                                            8, 9, 10, 11, 12, 13, 14, 15);
      if (ct + 1 < MT) {
        const __bf16* bp = wt_lds + (long long)((ct + 1) * 16 + laneLo) * K + kb;
        blo = *(const v8bf*)bp;
        bhi = *(const v8bf*)(bp + 8);
      }
      acc[ct] = __builtin_amdgcn_wmma_f32_16x16x32_bf16(
          false, afrag, false, bfrag, (short)0, acc[ct], false, false);
    }
  }

  // C layout: VGPR r, lanes 0-15 -> row r, col lane; lanes 16-31 -> row r+8.
  const int rbase = row0 + hiHalf * 8;
#pragma unroll
  for (int ct = 0; ct < MT; ++ct)
#pragma unroll
    for (int r = 0; r < 8; ++r)
      if (rbase + r < n)
        C[(long long)(rbase + r) * M + ct * 16 + laneLo] = acc[ct][r];
}

// ----------------------------- aggregation ---------------------------------
// One wave per edge; lane strides channels -> coalesced loads + atomics.

__global__ __launch_bounds__(256) void k_agg_edges(const float* __restrict__ h,
                                                   const long long* __restrict__ src,
                                                   const long long* __restrict__ dst,
                                                   const float* __restrict__ dinv,
                                                   float* __restrict__ out,
                                                   int E, int D) {
  int w    = (int)(((long long)blockIdx.x * blockDim.x + threadIdx.x) >> 5);
  int lane = threadIdx.x & 31;
  if (w >= E) return;
  int s = (int)src[w], t = (int)dst[w];
  float norm = dinv[s] * dinv[t];
#pragma unroll 4
  for (int c = lane; c < D; c += 32)
    atomicAdd(&out[(long long)t * D + c], h[(long long)s * D + c] * norm);
}

__global__ __launch_bounds__(256) void k_self_bias(const float* __restrict__ h,
                                                   const float* __restrict__ dinv,
                                                   const float* __restrict__ b,
                                                   float* __restrict__ out,
                                                   long long total, int D) {
  long long i = (long long)blockIdx.x * blockDim.x + threadIdx.x;
  if (i >= total) return;
  int r = (int)(i / D), c = (int)(i % D);
  float di = dinv[r];
  out[i] += h[i] * di * di + b[c];
}

// ----------------------------- batch-norm ----------------------------------

__global__ __launch_bounds__(128) void k_bn_stats(const float* __restrict__ a,
                                                  float* __restrict__ sum,
                                                  float* __restrict__ sumsq,
                                                  int N, int D) {
  int c = threadIdx.x;                               // blockDim.x == D == 128
  float s = 0.f, q = 0.f;
  for (int r = blockIdx.x; r < N; r += gridDim.x) {
    float v = a[(long long)r * D + c];
    s += v; q += v * v;
  }
  atomicAdd(&sum[c], s);
  atomicAdd(&sumsq[c], q);
}

__global__ __launch_bounds__(128) void k_bn_finalize(const float* __restrict__ sum,
                                                     const float* __restrict__ sumsq,
                                                     const float* __restrict__ g,
                                                     const float* __restrict__ be,
                                                     float* __restrict__ scale,
                                                     float* __restrict__ shift,
                                                     int N, int D) {
  int c = threadIdx.x;
  if (c >= D) return;
  float mu  = sum[c] / (float)N;
  float var = sumsq[c] / (float)N - mu * mu;
  float sc  = rsqrtf(var + BN_EPS) * g[c];
  scale[c] = sc;
  shift[c] = be[c] - mu * sc;
}

// BN apply + ReLU + down-convert to bf16 for the next WMMA GEMM input.
__global__ __launch_bounds__(256) void k_bn_apply_relu_bf(const float* __restrict__ a,
                                                          const float* __restrict__ scale,
                                                          const float* __restrict__ shift,
                                                          __bf16* __restrict__ outbf,
                                                          long long total, int D) {
  long long i = (long long)blockIdx.x * blockDim.x + threadIdx.x;
  if (i >= total) return;
  int c = (int)(i % D);
  float v = fmaxf(a[i] * scale[c] + shift[c], 0.0f);
  outbf[i] = (__bf16)v;
}

// ----------------------------- pooling -------------------------------------

__global__ __launch_bounds__(256) void k_pool(const float* __restrict__ a,
                                              const long long* __restrict__ batch,
                                              float* __restrict__ sums,
                                              float* __restrict__ cnt,
                                              int N, int D) {
  int w    = (int)(((long long)blockIdx.x * blockDim.x + threadIdx.x) >> 5);
  int lane = threadIdx.x & 31;
  if (w >= N) return;
  int g = (int)batch[w];
#pragma unroll 2
  for (int c = lane; c < D; c += 32)
    atomicAdd(&sums[g * D + c], a[(long long)w * D + c]);
  if (lane == 0) atomicAdd(&cnt[g], 1.0f);
}

__global__ __launch_bounds__(256) void k_pool_div(const float* __restrict__ sums,
                                                  const float* __restrict__ cnt,
                                                  float* __restrict__ out,
                                                  int total, int D) {
  int i = blockIdx.x * blockDim.x + threadIdx.x;
  if (i < total) out[i] = sums[i] / fmaxf(cnt[i / D], 1.0f);
}

// ----------------------------- launch --------------------------------------

extern "C" void kernel_launch(void* const* d_in, const int* in_sizes, int n_in,
                              void* d_out, int out_size, void* d_ws, size_t ws_size,
                              hipStream_t stream) {
  const float*     x     = (const float*)d_in[0];
  const long long* ei    = (const long long*)d_in[1];   // int64 per reference
  const long long* batch = (const long long*)d_in[2];   // int64 per reference
  const float* W1 = (const float*)d_in[3];  const float* b1  = (const float*)d_in[4];
  const float* W2 = (const float*)d_in[5];  const float* b2  = (const float*)d_in[6];
  const float* W3 = (const float*)d_in[7];  const float* b3  = (const float*)d_in[8];
  const float* g1 = (const float*)d_in[9];  const float* be1 = (const float*)d_in[10];
  const float* g2 = (const float*)d_in[11]; const float* be2 = (const float*)d_in[12];

  const int N = in_sizes[0] / D_IN;
  const int E = in_sizes[1] / 2;
  const long long* src = ei;
  const long long* dst = ei + E;

  // ---- workspace carve-out (256B aligned) ----
  char* ws = (char*)d_ws;
  size_t off = 0;
  auto carve = [&](size_t bytes) -> void* {
    void* p = ws + off;
    off = (off + bytes + 255) & ~(size_t)255;
    return p;
  };
  float*  dinv   = (float*)carve((size_t)N * 4);
  __bf16* inbf   = (__bf16*)carve((size_t)N * D_IN * 2);   // layer input (bf16)
  __bf16* WT1    = (__bf16*)carve((size_t)D_HID * D_IN * 2);
  __bf16* WT2    = (__bf16*)carve((size_t)D_HID * D_HID * 2);
  __bf16* WT3    = (__bf16*)carve((size_t)D_OUT * D_HID * 2);
  float*  H      = (float*)carve((size_t)N * D_HID * 4);   // GEMM output
  float*  AGG    = (float*)carve((size_t)N * D_HID * 4);   // aggregation output
  float*  bnsum  = (float*)carve(256 * 4);                 // sum(128) + sumsq(128)
  float*  bnsq   = bnsum + 128;
  float*  bnscale= (float*)carve(128 * 4);
  float*  bnshift= (float*)carve(128 * 4);
  float*  pools  = (float*)carve((NUM_GRAPHS * D_OUT + NUM_GRAPHS) * 4);
  float*  pcnt   = pools + NUM_GRAPHS * D_OUT;
  (void)ws_size; (void)n_in; (void)out_size;

  const int nRowTiles = (N + 15) / 16;
  const int gemmBlocks = (nRowTiles + 7) / 8;
  const int aggBlocks  = (E + 7) / 8;

  // ---- degrees & symmetric norm ----
  hipMemsetAsync(dinv, 0, (size_t)N * 4, stream);
  k_degree<<<(E + 255) / 256, 256, 0, stream>>>(dst, dinv, E);
  k_dinv<<<(N + 255) / 256, 256, 0, stream>>>(dinv, N);

  // ---- precision staging ----
  {
    long long tot = (long long)N * D_IN;
    k_to_bf16<<<(int)((tot + 255) / 256), 256, 0, stream>>>(x, inbf, tot);
  }
  k_transpose_bf16<<<(D_IN * D_HID + 255) / 256, 256, 0, stream>>>(W1, WT1, D_IN, D_HID);
  k_transpose_bf16<<<(D_HID * D_HID + 255) / 256, 256, 0, stream>>>(W2, WT2, D_HID, D_HID);
  k_transpose_bf16<<<(D_HID * D_OUT + 255) / 256, 256, 0, stream>>>(W3, WT3, D_HID, D_OUT);

  // ======================= layer 1 =======================
  k_gemm_wmma<D_IN, D_HID><<<gemmBlocks, 256, D_IN * D_HID * 2, stream>>>(
      inbf, WT1, H, N, nRowTiles);
  hipMemsetAsync(AGG, 0, (size_t)N * D_HID * 4, stream);
  k_agg_edges<<<aggBlocks, 256, 0, stream>>>(H, src, dst, dinv, AGG, E, D_HID);
  {
    long long tot = (long long)N * D_HID;
    k_self_bias<<<(int)((tot + 255) / 256), 256, 0, stream>>>(H, dinv, b1, AGG, tot, D_HID);
    hipMemsetAsync(bnsum, 0, 256 * 4, stream);
    k_bn_stats<<<256, 128, 0, stream>>>(AGG, bnsum, bnsq, N, D_HID);
    k_bn_finalize<<<1, 128, 0, stream>>>(bnsum, bnsq, g1, be1, bnscale, bnshift, N, D_HID);
    k_bn_apply_relu_bf<<<(int)((tot + 255) / 256), 256, 0, stream>>>(
        AGG, bnscale, bnshift, inbf, tot, D_HID);
  }

  // ======================= layer 2 =======================
  k_gemm_wmma<D_HID, D_HID><<<gemmBlocks, 256, D_HID * D_HID * 2, stream>>>(
      inbf, WT2, H, N, nRowTiles);
  hipMemsetAsync(AGG, 0, (size_t)N * D_HID * 4, stream);
  k_agg_edges<<<aggBlocks, 256, 0, stream>>>(H, src, dst, dinv, AGG, E, D_HID);
  {
    long long tot = (long long)N * D_HID;
    k_self_bias<<<(int)((tot + 255) / 256), 256, 0, stream>>>(H, dinv, b2, AGG, tot, D_HID);
    hipMemsetAsync(bnsum, 0, 256 * 4, stream);
    k_bn_stats<<<256, 128, 0, stream>>>(AGG, bnsum, bnsq, N, D_HID);
    k_bn_finalize<<<1, 128, 0, stream>>>(bnsum, bnsq, g2, be2, bnscale, bnshift, N, D_HID);
    k_bn_apply_relu_bf<<<(int)((tot + 255) / 256), 256, 0, stream>>>(
        AGG, bnscale, bnshift, inbf, tot, D_HID);
  }

  // ======================= layer 3 =======================
  k_gemm_wmma<D_HID, D_OUT><<<gemmBlocks, 256, D_HID * D_OUT * 2, stream>>>(
      inbf, WT3, H, N, nRowTiles);
  hipMemsetAsync(AGG, 0, (size_t)N * D_OUT * 4, stream);
  k_agg_edges<<<aggBlocks, 256, 0, stream>>>(H, src, dst, dinv, AGG, E, D_OUT);
  {
    long long tot = (long long)N * D_OUT;
    k_self_bias<<<(int)((tot + 255) / 256), 256, 0, stream>>>(H, dinv, b3, AGG, tot, D_OUT);
  }

  // ======================= global mean pool =======================
  hipMemsetAsync(pools, 0, (NUM_GRAPHS * D_OUT + NUM_GRAPHS) * 4, stream);
  k_pool<<<(N + 7) / 8, 256, 0, stream>>>(AGG, batch, pools, pcnt, N, D_OUT);
  k_pool_div<<<(NUM_GRAPHS * D_OUT + 255) / 256, 256, 0, stream>>>(
      pools, pcnt, (float*)d_out, NUM_GRAPHS * D_OUT, D_OUT);
}
